// SSDMixer_3745211482478
// MI455X (gfx1250) — compile-verified, matrix operand order checked by reading, not run on hardware
//
#include <hip/hip_runtime.h>
#include <hip/hip_bf16.h>

#define L_SEQ    2048
#define D_MODEL  2048
#define D_INNER  4096
#define NHEADS   64
#define HEADDIM  64
#define D_STATE  128
#define D_CONV   4
#define CONV_DIM 4352           // D_INNER + 2*D_STATE
#define D_IN_PROJ 8512          // 2*D_INNER + 2*D_STATE + NHEADS
#define CHUNK    256
#define NCHUNK   8
#define XT_LD    272            // 256 + 16 pad (keeps 32B alignment, spreads LDS banks)
#define GEMM_LD  80             // 64 + 16 pad, row stride 160B (multiple of 32B)

typedef __attribute__((ext_vector_type(16))) __bf16 v16bf;
typedef __attribute__((ext_vector_type(8)))  __bf16 v8bf;
typedef __attribute__((ext_vector_type(8)))  float  v8f;

// gfx1250 async global->LDS copy path (ASYNCcnt), guarded so the file always compiles.
#if defined(__has_builtin)
#if __has_builtin(__builtin_amdgcn_global_load_async_to_lds_b128) && \
    __has_builtin(__builtin_amdgcn_s_wait_asynccnt)
#define ASYNC_LDS 1
#endif
#endif
#ifndef ASYNC_LDS
#define ASYNC_LDS 0
#endif

#if ASYNC_LDS
typedef int v4i_g __attribute__((vector_size(16)));                 // matches builtin's pointee type
typedef __attribute__((address_space(1))) v4i_g* as1_v4i_p;         // global (__device__) int4*
typedef __attribute__((address_space(3))) v4i_g* as3_v4i_p;         // LDS (__shared__) int4*
#endif

// copy 16 bytes global -> LDS
static __device__ inline void cp16(const __bf16* g, __bf16* l) {
#if ASYNC_LDS
  __builtin_amdgcn_global_load_async_to_lds_b128(
      (as1_v4i_p)(void*)g, (as3_v4i_p)(void*)l, 0, 0);
#else
  *(v8bf*)l = *(const v8bf*)g;
#endif
}

static __device__ inline void stage_wait() {
#if ASYNC_LDS
  __builtin_amdgcn_s_wait_asynccnt(0);
#endif
  __syncthreads();
}

static __device__ inline v8f vzero() {
  v8f z = {0.f,0.f,0.f,0.f,0.f,0.f,0.f,0.f};
  return z;
}

static __device__ inline v8f wmma_bf16(v16bf a, v16bf b, v8f c) {
  // D = A(16x32 bf16) x B(32x16 bf16) + C(f32)
  return __builtin_amdgcn_wmma_f32_16x16x32_bf16(false, a, false, b, (short)0, c, false, false);
}

// A/B fragment from row-major [rows][ld] bf16 source, K contiguous.
// lane L: row = L&15, k = k0 + (L>>4)*16 .. +15  (contiguous k order, used for both A and B)
static __device__ inline v16bf load_frag_bf16(const __bf16* __restrict__ base, int ld, int k0) {
  int lane = threadIdx.x & 31;
  int rr = lane & 15;
  int kk = (lane >> 4) << 4;
  return *(const v16bf*)(base + (size_t)rr * ld + k0 + kk);
}

// Same fragment but converting from f32 source on the fly.
static __device__ inline v16bf load_frag_f32(const float* __restrict__ base, int ld, int k0) {
  int lane = threadIdx.x & 31;
  int rr = lane & 15;
  int kk = (lane >> 4) << 4;
  const float* p = base + (size_t)rr * ld + k0 + kk;
  v16bf v;
#pragma unroll
  for (int i = 0; i < 16; ++i) v[i] = (__bf16)p[i];
  return v;
}

// Store 16x16 f32 D tile: VGPR r -> M = r + (lane>=16 ? 8 : 0), N = lane&15
static __device__ inline void store_tile(float* __restrict__ base, int ld, v8f d) {
  int lane = threadIdx.x & 31;
  int n  = lane & 15;
  int mb = (lane >> 4) * 8;
#pragma unroll
  for (int r = 0; r < 8; ++r) base[(size_t)(mb + r) * ld + n] = d[r];
}

// ---------------------------------------------------------------- converts
__global__ __launch_bounds__(256) void k_f32_to_bf16(const float* __restrict__ src,
                                                     __bf16* __restrict__ dst, int n) {
  int i = blockIdx.x * 256 + threadIdx.x;
  if (i < n) dst[i] = (__bf16)src[i];
}

// ---------------------------------------------------------------- NT GEMM: C[MxN] = A[MxK] * B[NxK]^T
// Per block: 128(M) x 64(N) tile, 8 waves (wave32). A/B K-tiles staged through
// LDS with double-buffered async global->LDS copies so the next tile's DMA
// overlaps the current tile's WMMAs; B tile is fetched once per block instead
// of once per wave.
__global__ __launch_bounds__(256) void k_gemm_nt_bf16(const __bf16* __restrict__ A,
                                                      const __bf16* __restrict__ B,
                                                      float* __restrict__ C,
                                                      int M, int N, int K) {
  __shared__ __attribute__((aligned(32))) __bf16 Abuf[2][128][GEMM_LD];
  __shared__ __attribute__((aligned(32))) __bf16 Bbuf[2][64][GEMM_LD];
  int t = threadIdx.x;
  int wave = t >> 5, lane = t & 31;
  int r = lane & 15, khalf = (lane >> 4) << 4;
  int m_blk = blockIdx.y * 128;
  int n0 = blockIdx.x * 64;
  const __bf16* Ag = A + (size_t)m_blk * K;
  const __bf16* Bg = B + (size_t)n0 * K;

  // stage one 64-wide K tile (A: 128x64, B: 64x64) into buffer `buf`
  auto stage = [&](int k0, int buf) {
#pragma unroll
    for (int i = 0; i < 4; ++i) {               // A: 1024 chunks of 16B
      int cid = t + i * 256;
      int row = cid >> 3, ck = (cid & 7) * 8;
      cp16(Ag + (size_t)row * K + k0 + ck, &Abuf[buf][row][ck]);
    }
#pragma unroll
    for (int i = 0; i < 2; ++i) {               // B: 512 chunks of 16B
      int cid = t + i * 256;
      int row = cid >> 3, ck = (cid & 7) * 8;
      cp16(Bg + (size_t)row * K + k0 + ck, &Bbuf[buf][row][ck]);
    }
    __builtin_prefetch(Ag + (size_t)(t >> 1) * K + k0 + 128, 0, 1); // next-next tile hint
  };

  v8f acc[4] = {vzero(), vzero(), vzero(), vzero()};
  int nk = K >> 6;
  stage(0, 0);
  stage_wait();
  int buf = 0;
  for (int kt = 0; kt < nk; ++kt) {
    if (kt + 1 < nk) stage((kt + 1) << 6, buf ^ 1);
#pragma unroll
    for (int ks = 0; ks < 64; ks += 32) {
      v16bf a = *(const v16bf*)&Abuf[buf][wave * 16 + r][ks + khalf];
#pragma unroll
      for (int j = 0; j < 4; ++j) {
        v16bf b = *(const v16bf*)&Bbuf[buf][j * 16 + r][ks + khalf];
        acc[j] = wmma_bf16(a, b, acc[j]);
      }
    }
    stage_wait();            // stage of buf^1 done + all waves done reading buf
    buf ^= 1;
  }
#pragma unroll
  for (int j = 0; j < 4; ++j)
    store_tile(C + (size_t)(m_blk + wave * 16) * N + n0 + 16 * j, N, acc[j]);
}

// ---------------------------------------------------------------- depthwise causal conv + silu
__global__ __launch_bounds__(256) void k_conv(const float* __restrict__ zx,
                                              const float* __restrict__ Wc,
                                              const float* __restrict__ bc,
                                              float* __restrict__ xBCc) {
  int idx = blockIdx.x * 256 + threadIdx.x;
  if (idx >= L_SEQ * CONV_DIM) return;
  int l = idx / CONV_DIM, c = idx % CONV_DIM;
  float acc = bc[c];
#pragma unroll
  for (int j = 0; j < D_CONV; ++j) {
    int tt = l - (D_CONV - 1) + j;
    if (tt >= 0) acc += Wc[c * D_CONV + j] * zx[(size_t)tt * D_IN_PROJ + D_INNER + c];
  }
  xBCc[idx] = acc / (1.f + __expf(-acc));
}

// ---------------------------------------------------------------- per (chunk,head) cumsum of dt*A
__global__ __launch_bounds__(256) void k_scan(const float* __restrict__ zx,
                                              const float* __restrict__ dt_bias,
                                              const float* __restrict__ A_log,
                                              float* __restrict__ cs,
                                              float* __restrict__ T) {
  int h = blockIdx.x, c = blockIdx.y, l = threadIdx.x;
  __shared__ float sh[CHUNK];
  float raw = zx[(size_t)(c * CHUNK + l) * D_IN_PROJ + (D_INNER + CONV_DIM) + h] + dt_bias[h];
  float dtv = (raw > 20.f) ? raw : log1pf(__expf(raw));       // softplus
  sh[l] = -__expf(A_log[h]) * dtv;
  __syncthreads();
  for (int off = 1; off < CHUNK; off <<= 1) {                 // inclusive scan
    float t = (l >= off) ? sh[l - off] : 0.f;
    __syncthreads();
    sh[l] += t;
    __syncthreads();
  }
  cs[((size_t)c * NHEADS + h) * CHUNK + l] = sh[l];
  if (l == CHUNK - 1) T[c * NHEADS + h] = sh[l];
}

// ---------------------------------------------------------------- G[c][l][s] = C[l]·B[s]  (lower-tri tiles)
__global__ __launch_bounds__(256) void k_gscore(const float* __restrict__ xBCc,
                                                float* __restrict__ G) {
  int c = blockIdx.x;
  int wave = threadIdx.x >> 5;
  const float* Cb = xBCc + (size_t)c * CHUNK * CONV_DIM + (D_INNER + D_STATE);
  const float* Bb = xBCc + (size_t)c * CHUNK * CONV_DIM + D_INNER;
  for (int t = wave; t < 256; t += 8) {
    int lt = t >> 4, st = t & 15;
    if (st > lt) continue;                                    // fully masked tiles
    v8f acc = vzero();
    for (int k0 = 0; k0 < D_STATE; k0 += 32) {
      v16bf a = load_frag_f32(Cb + (size_t)(lt * 16) * CONV_DIM, CONV_DIM, k0);
      v16bf b = load_frag_f32(Bb + (size_t)(st * 16) * CONV_DIM, CONV_DIM, k0);
      acc = wmma_bf16(a, b, acc);
    }
    store_tile(G + (size_t)c * CHUNK * CHUNK + (size_t)lt * 16 * CHUNK + st * 16, CHUNK, acc);
  }
}

// ---------------------------------------------------------------- Y_diag + D*X, and chunk states
__global__ __launch_bounds__(256) void k_ssd_diag_states(const float* __restrict__ xBCc,
                                                         const float* __restrict__ cs,
                                                         const float* __restrict__ G,
                                                         const float* __restrict__ Dp,
                                                         float* __restrict__ Y,
                                                         float* __restrict__ states) {
  int h = blockIdx.x, c = blockIdx.y;
  __shared__ __attribute__((aligned(32))) __bf16 Xt[HEADDIM][XT_LD];   // X^T  [p][l]
  __shared__ __attribute__((aligned(32))) __bf16 Btd[D_STATE][XT_LD];  // (dec*B)^T [n][l]
  __shared__ float csh[CHUNK];
  int t = threadIdx.x;
  csh[t] = cs[((size_t)c * NHEADS + h) * CHUNK + t];
  __syncthreads();
  float cl_last = csh[CHUNK - 1];
  for (int i = t; i < HEADDIM * CHUNK; i += 256) {
    int p = i >> 8, l = i & 255;
    Xt[p][l] = (__bf16)xBCc[(size_t)(c * CHUNK + l) * CONV_DIM + h * HEADDIM + p];
  }
  for (int i = t; i < D_STATE * CHUNK; i += 256) {
    int n = i >> 8, l = i & 255;
    float bv = xBCc[(size_t)(c * CHUNK + l) * CONV_DIM + D_INNER + n];
    Btd[n][l] = (__bf16)(bv * __expf(cl_last - csh[l]));
  }
  __syncthreads();

  int wave = t >> 5, lane = t & 31;
  int r = lane & 15, khalf = (lane >> 4) << 4, halfsel = (lane >> 4) * 8;
  float dph = Dp[h];

  // Phase A: Y_diag[l,p] = sum_s (G[l,s]*exp(cs_l-cs_s))[s<=l] * X[s,p]   (+ D*X)
  for (int wi = 0; wi < 2; ++wi) {
    int lt = wave + wi * 8;
    int l = lt * 16 + r;
    float cl = csh[l];
    const float* grow = G + (size_t)c * CHUNK * CHUNK + (size_t)l * CHUNK;
    v8f acc[4] = {vzero(), vzero(), vzero(), vzero()};
    for (int k0 = 0; k0 <= lt * 16; k0 += 32) {
      v16bf a;
#pragma unroll
      for (int i2 = 0; i2 < 16; ++i2) {
        int s = k0 + khalf + i2;
        float v = (s <= l) ? grow[s] * __expf(cl - csh[s]) : 0.f;
        a[i2] = (__bf16)v;
      }
#pragma unroll
      for (int j = 0; j < 4; ++j) {
        v16bf b = *(const v16bf*)&Xt[j * 16 + r][k0 + khalf];
        acc[j] = wmma_bf16(a, b, acc[j]);
      }
    }
#pragma unroll
    for (int j = 0; j < 4; ++j) {
      int p = j * 16 + r;
#pragma unroll
      for (int rr = 0; rr < 8; ++rr) {
        int lo = lt * 16 + halfsel + rr;
        float xv = xBCc[(size_t)(c * CHUNK + lo) * CONV_DIM + h * HEADDIM + p];
        Y[(size_t)(c * CHUNK + lo) * D_INNER + h * HEADDIM + p] = acc[j][rr] + dph * xv;
      }
    }
  }

  // Phase B: states[p,n] = sum_l X[l,p] * (dec[l]*B[l,n])
  float* sp = states + (size_t)(c * NHEADS + h) * HEADDIM * D_STATE;
  for (int tb = wave * 4; tb < wave * 4 + 4; ++tb) {
    int pt = tb & 3, nt = tb >> 2;
    v8f acc = vzero();
    for (int k0 = 0; k0 < CHUNK; k0 += 32) {
      v16bf a = *(const v16bf*)&Xt[pt * 16 + r][k0 + khalf];
      v16bf b = *(const v16bf*)&Btd[nt * 16 + r][k0 + khalf];
      acc = wmma_bf16(a, b, acc);
    }
#pragma unroll
    for (int rr = 0; rr < 8; ++rr)
      sp[(size_t)(pt * 16 + halfsel + rr) * D_STATE + nt * 16 + r] = acc[rr];
  }
}

// ---------------------------------------------------------------- inter-chunk recurrence: S_in[z+1]=exp(T_z)*S_in[z]+states[z]
__global__ __launch_bounds__(256) void k_chain(const float* __restrict__ states,
                                               const float* __restrict__ T,
                                               float* __restrict__ S_in) {
  int h = blockIdx.x, t = threadIdx.x;
  float acc[32];
#pragma unroll
  for (int i = 0; i < 32; ++i) acc[i] = 0.f;
  for (int z = 0; z < NCHUNK; ++z) {
    const float* sz = states + (size_t)(z * NHEADS + h) * HEADDIM * D_STATE;
    float* dz = S_in + (size_t)(z * NHEADS + h) * HEADDIM * D_STATE;
    float eT = __expf(T[z * NHEADS + h]);
#pragma unroll
    for (int i = 0; i < 32; ++i) {
      int idx = t + i * 256;
      dz[idx] = acc[i];
      acc[i] = eT * acc[i] + sz[idx];
    }
  }
}

// ---------------------------------------------------------------- Y += exp(cs_l) * C[l,:] @ S_in[p,:]^T
__global__ __launch_bounds__(256) void k_yoff(const float* __restrict__ xBCc,
                                              const float* __restrict__ cs,
                                              const float* __restrict__ S_in,
                                              float* __restrict__ Y) {
  int h = blockIdx.x, c = blockIdx.y;
  __shared__ float csh[CHUNK];
  int t = threadIdx.x;
  csh[t] = cs[((size_t)c * NHEADS + h) * CHUNK + t];
  __syncthreads();
  int wave = t >> 5, lane = t & 31;
  int r = lane & 15, halfsel = (lane >> 4) * 8;
  const float* Cb = xBCc + (size_t)c * CHUNK * CONV_DIM + (D_INNER + D_STATE);
  const float* Sb = S_in + (size_t)(c * NHEADS + h) * HEADDIM * D_STATE;
  for (int wi = 0; wi < 2; ++wi) {
    int lt = wave + wi * 8;
    v8f acc[4] = {vzero(), vzero(), vzero(), vzero()};
    for (int k0 = 0; k0 < D_STATE; k0 += 32) {
      v16bf a = load_frag_f32(Cb + (size_t)(lt * 16) * CONV_DIM, CONV_DIM, k0);
#pragma unroll
      for (int j = 0; j < 4; ++j) {
        v16bf b = load_frag_f32(Sb + (size_t)(j * 16) * D_STATE, D_STATE, k0);
        acc[j] = wmma_bf16(a, b, acc[j]);
      }
    }
#pragma unroll
    for (int j = 0; j < 4; ++j) {
      int p = j * 16 + r;
#pragma unroll
      for (int rr = 0; rr < 8; ++rr) {
        int lo = lt * 16 + halfsel + rr;
        size_t yi = (size_t)(c * CHUNK + lo) * D_INNER + h * HEADDIM + p;
        Y[yi] += __expf(csh[lo]) * acc[j][rr];
      }
    }
  }
}

// ---------------------------------------------------------------- gate (silu(z)) + RMSNorm -> bf16
__global__ __launch_bounds__(256) void k_gate_norm(const float* __restrict__ Y,
                                                   const float* __restrict__ zx,
                                                   const float* __restrict__ norm_w,
                                                   __bf16* __restrict__ ybf) {
  int l = blockIdx.x, t = threadIdx.x;
  __shared__ float red[256];
  float vals[16];
  float ss = 0.f;
#pragma unroll
  for (int i = 0; i < 16; ++i) {
    int d = t + i * 256;
    float zv = zx[(size_t)l * D_IN_PROJ + d];
    float v = Y[(size_t)l * D_INNER + d] * (zv / (1.f + __expf(-zv)));
    vals[i] = v;
    ss += v * v;
  }
  red[t] = ss;
  __syncthreads();
  for (int s = 128; s > 0; s >>= 1) {
    if (t < s) red[t] += red[t + s];
    __syncthreads();
  }
  float rms = rsqrtf(red[0] / (float)D_INNER + 1e-5f);
#pragma unroll
  for (int i = 0; i < 16; ++i) {
    int d = t + i * 256;
    ybf[(size_t)l * D_INNER + d] = (__bf16)(vals[i] * rms * norm_w[d]);
  }
}

// ================================================================ host
extern "C" void kernel_launch(void* const* d_in, const int* in_sizes, int n_in,
                              void* d_out, int out_size, void* d_ws, size_t ws_size,
                              hipStream_t stream) {
  const float* u       = (const float*)d_in[0];
  const float* W_in    = (const float*)d_in[1];
  const float* W_conv  = (const float*)d_in[2];
  const float* b_conv  = (const float*)d_in[3];
  const float* dt_bias = (const float*)d_in[4];
  const float* A_log   = (const float*)d_in[5];
  const float* Dp      = (const float*)d_in[6];
  const float* norm_w  = (const float*)d_in[7];
  const float* W_out   = (const float*)d_in[8];
  float* out = (float*)d_out;

  char* ws = (char*)d_ws;
  size_t off = 0;
  auto alloc = [&](size_t bytes) -> char* {
    char* p = ws + off;
    off += (bytes + 255) & ~(size_t)255;
    return p;
  };
  __bf16* u_bf    = (__bf16*)alloc((size_t)L_SEQ * D_MODEL * 2);
  __bf16* Win_bf  = (__bf16*)alloc((size_t)D_IN_PROJ * D_MODEL * 2);
  __bf16* Wout_bf = (__bf16*)alloc((size_t)D_MODEL * D_INNER * 2);
  float*  zx      = (float*)alloc((size_t)L_SEQ * D_IN_PROJ * 4);
  float*  xBCc    = (float*)alloc((size_t)L_SEQ * CONV_DIM * 4);
  float*  cs      = (float*)alloc((size_t)NCHUNK * NHEADS * CHUNK * 4);
  float*  T       = (float*)alloc((size_t)NCHUNK * NHEADS * 4);
  float*  G       = (float*)alloc((size_t)NCHUNK * CHUNK * CHUNK * 4);
  float*  states  = (float*)alloc((size_t)NCHUNK * NHEADS * HEADDIM * D_STATE * 4);
  float*  S_in    = (float*)alloc((size_t)NCHUNK * NHEADS * HEADDIM * D_STATE * 4);
  float*  Y       = (float*)alloc((size_t)L_SEQ * D_INNER * 4);
  __bf16* y_bf    = (__bf16*)alloc((size_t)L_SEQ * D_INNER * 2);
  (void)ws_size; (void)in_sizes; (void)n_in; (void)out_size;

  // f32 -> bf16 operand conversion (once per launch)
  {
    int n1 = L_SEQ * D_MODEL;
    k_f32_to_bf16<<<(n1 + 255) / 256, 256, 0, stream>>>(u, u_bf, n1);
    int n2 = D_IN_PROJ * D_MODEL;
    k_f32_to_bf16<<<(n2 + 255) / 256, 256, 0, stream>>>(W_in, Win_bf, n2);
    int n3 = D_MODEL * D_INNER;
    k_f32_to_bf16<<<(n3 + 255) / 256, 256, 0, stream>>>(W_out, Wout_bf, n3);
  }
  // in_proj: zx[2048 x 8512]
  k_gemm_nt_bf16<<<dim3(D_IN_PROJ / 64, L_SEQ / 128), 256, 0, stream>>>(
      u_bf, Win_bf, zx, L_SEQ, D_IN_PROJ, D_MODEL);
  // depthwise causal conv + silu
  k_conv<<<(L_SEQ * CONV_DIM + 255) / 256, 256, 0, stream>>>(zx, W_conv, b_conv, xBCc);
  // dt*A cumsums per (chunk, head)
  k_scan<<<dim3(NHEADS, NCHUNK), CHUNK, 0, stream>>>(zx, dt_bias, A_log, cs, T);
  // C·B^T per chunk
  k_gscore<<<NCHUNK, 256, 0, stream>>>(xBCc, G);
  // intra-chunk diagonal output + chunk states
  k_ssd_diag_states<<<dim3(NHEADS, NCHUNK), 256, 0, stream>>>(xBCc, cs, G, Dp, Y, states);
  // inter-chunk state recurrence
  k_chain<<<NHEADS, 256, 0, stream>>>(states, T, S_in);
  // off-diagonal correction
  k_yoff<<<dim3(NHEADS, NCHUNK), 256, 0, stream>>>(xBCc, cs, S_in, Y);
  // gate + RMSNorm -> bf16
  k_gate_norm<<<L_SEQ, 256, 0, stream>>>(Y, zx, norm_w, y_bf);
  // out_proj -> f32 output
  k_gemm_nt_bf16<<<dim3(D_MODEL / 64, L_SEQ / 128), 256, 0, stream>>>(
      y_bf, Wout_bf, out, L_SEQ, D_MODEL, D_INNER);
}
